// PatchBackbone_79946521247944
// MI455X (gfx1250) — compile-verified, matrix operand order checked by reading, not run on hardware
//
#include <hip/hip_runtime.h>
#include <hip/hip_bf16.h>
#include <math.h>

typedef __attribute__((ext_vector_type(16))) _Float16 v16h;
typedef __attribute__((ext_vector_type(8)))  float    v8f;
typedef __attribute__((ext_vector_type(4)))  unsigned int v4u;
typedef __attribute__((ext_vector_type(4)))  int v4i;
typedef __attribute__((ext_vector_type(8)))  int v8i;

#define WAVE ((int)(threadIdx.x >> 5))
#define LANE ((int)(threadIdx.x & 31))

constexpr int Bsz = 32, Pp = 128, Ll = 64, DINc = 16, NHc = 8, NLc = 2;
constexpr int DPpad = 96;                 // DP=80 padded to 96 (3 x K=32 tiles)
constexpr int Npatch = Bsz * Pp;          // 4096
constexpr float DToverTAU = 300.0f / 300.0f;

// ---- workspace layout (offsets in halves unless noted) ----
constexpr size_t W_INP = 0;                           // [512][96]
constexpr size_t W_WQ  = W_INP + (size_t)512 * 96;
constexpr size_t W_WK  = W_WQ  + (size_t)512 * 512;
constexpr size_t W_WV  = W_WK  + (size_t)512 * 512;
constexpr size_t W_WO  = W_WV  + (size_t)512 * 512;
constexpr size_t W_F1  = W_WO  + (size_t)512 * 512;
constexpr size_t W_F2  = W_F1  + (size_t)512 * 512;
constexpr size_t W_G   = W_F2  + (size_t)512 * 512;   // + (li*4 + {q,k,v,o})*512*512
constexpr size_t TOK_OFF = W_G + (size_t)8 * 512 * 512;          // tokens f16 [N][512]
constexpr size_t Z_OFF   = TOK_OFF + (size_t)Npatch * 512;       // z f16 [N][512]
constexpr size_t PEX_OFFB = (Z_OFF + (size_t)Npatch * 512) * 2;  // byte offset of pexists f32[N]

// =============== WMMA fragment helpers (wave32, ISA 7.12.2 layouts) ===============
__device__ __forceinline__ v16h ldA16x32(const _Float16* src, int row0, int col0, int ld) {
  const int m = LANE & 15, koff = (LANE >> 4) << 3;
  const _Float16* p = src + (size_t)(row0 + m) * ld + col0 + koff;
  v16h a;
#pragma unroll
  for (int e = 0; e < 8; ++e) { a[e] = p[e]; a[e + 8] = p[e + 16]; }
  return a;
}
// B[k][n] = src[n][k]  (weights stored [out][in], or K^T for Q@K^T)
__device__ __forceinline__ v16h ldB_nk(const _Float16* src, int n0, int k0, int ld) {
  const int n = LANE & 15, koff = (LANE >> 4) << 3;
  const _Float16* p = src + (size_t)(n0 + n) * ld + k0 + koff;
  v16h b;
#pragma unroll
  for (int e = 0; e < 8; ++e) { b[e] = p[e]; b[e + 8] = p[e + 16]; }
  return b;
}
// B[k][n] = src[k][n]  (e.g. V matrix for P@V)
__device__ __forceinline__ v16h ldB_kn(const _Float16* src, int k0, int n0, int ld) {
  const int n = LANE & 15, koff = (LANE >> 4) << 3;
  v16h b;
#pragma unroll
  for (int e = 0; e < 8; ++e) {
    b[e]     = src[(size_t)(k0 + koff + e) * ld + n0 + n];
    b[e + 8] = src[(size_t)(k0 + koff + 16 + e) * ld + n0 + n];
  }
  return b;
}
__device__ __forceinline__ v8f ldC_f16(const _Float16* src, int row0, int col0, int ld) {
  const int n = LANE & 15, moff = (LANE >> 4) << 3;
  v8f c;
#pragma unroll
  for (int r = 0; r < 8; ++r) c[r] = (float)src[(size_t)(row0 + moff + r) * ld + col0 + n];
  return c;
}
__device__ __forceinline__ void stD_f16(_Float16* dst, v8f d, int row0, int col0, int ld) {
  const int n = LANE & 15, moff = (LANE >> 4) << 3;
#pragma unroll
  for (int r = 0; r < 8; ++r) dst[(size_t)(row0 + moff + r) * ld + col0 + n] = (_Float16)d[r];
}
__device__ __forceinline__ v8f wmma_f16(v16h a, v16h b, v8f c) {
  return __builtin_amdgcn_wmma_f32_16x16x32_f16(false, a, false, b, (short)0, c, false, false);
}
__device__ __forceinline__ v8f bcast8(float v) { v8f c;
#pragma unroll
  for (int r = 0; r < 8; ++r) c[r] = v; return c; }

// =============== Tensor Data Mover: stage a 2-D f16 tile (tile_w x tile_h, row stride) into LDS ===============
// D# per CDNA5 ISA: group0 {count, lds_addr, global_addr, type=2},
// group1 {data_size=2B, tensor_dim0=stride, tensor_dim1=tile_h, tile_dim0=tile_w, tile_dim1=tile_h,
//         tensor_dim0_stride=stride}. Groups 2/3 zero (<=2D). Tracked on TENSORcnt.
__device__ __forceinline__ void tdm_stage_f16(const _Float16* gsrc, _Float16* lds_dst,
                                              unsigned tile_w, unsigned tile_h, unsigned stride) {
  unsigned long long ga = (unsigned long long)(const void*)gsrc;
  unsigned ldsa = (unsigned)(unsigned long long)(void*)lds_dst;  // low 32 bits = LDS byte offset
  v4u g0;
  g0[0] = 1u;                                     // count=1, is_restore=0, gather=0
  g0[1] = ldsa;                                   // lds_addr
  g0[2] = (unsigned)ga;                           // global_addr[31:0]
  g0[3] = (unsigned)((ga >> 32) & 0x01FFFFFFu) | (2u << 30);  // global_addr[56:32] | type=2
  v8i g1;
  g1[0] = (int)(1u << 16);                        // workgroup_mask=0, data_size=1 (2 bytes)
  g1[1] = (int)((stride & 0xFFFFu) << 16);        // tensor_dim0[15:0]
  g1[2] = (int)(((stride >> 16) & 0xFFFFu) | ((tile_h & 0xFFFFu) << 16)); // dim0 hi | dim1 lo
  g1[3] = (int)((tile_w & 0xFFFFu) << 16);        // tensor_dim1 hi (0) | tile_dim0
  g1[4] = (int)(tile_h & 0xFFFFu);                // tile_dim1 (tile_dim2 = 0)
  g1[5] = (int)stride;                            // tensor_dim0_stride[31:0]
  g1[6] = 0;                                      // dim0_stride hi | dim1_stride lo (unused, 2D)
  g1[7] = 0;
  v4i gz = {0, 0, 0, 0};
#if defined(__clang_major__) && (__clang_major__ >= 23)
  v8i gz8 = {0, 0, 0, 0, 0, 0, 0, 0};
  __builtin_amdgcn_tensor_load_to_lds(g0, g1, gz, gz, gz8, 0);
#else
  __builtin_amdgcn_tensor_load_to_lds(g0, g1, gz, gz, 0);
#endif
}
__device__ __forceinline__ void tensor_wait0() {
  __builtin_amdgcn_s_wait_tensorcnt((short)0);
}

// =============== weight pack fp32 -> f16 (zero-pads cols to dld) ===============
__global__ void pack_w(const float* __restrict__ src, _Float16* __restrict__ dst,
                       int rows, int cols, int dld) {
  int i = blockIdx.x * blockDim.x + threadIdx.x;
  if (i >= rows * dld) return;
  int r = i / dld, c = i % dld;
  dst[i] = (c < cols) ? (_Float16)src[(size_t)r * cols + c] : (_Float16)0.0f;
}

// =============== fused per-patch encoder: pts -> h0 -> MHA -> FFN -> pooled token ===============
__global__ void __launch_bounds__(256)
patch_kernel(const float* __restrict__ x, const float* __restrict__ tt, const int* __restrict__ mask,
             const float* tsw, const float* tsb, const float* tpw, const float* tpb,
             const float* inp_b, const float* bq, const float* bk, const float* bv, const float* bo,
             const float* fb1, const float* fb2,
             const float* ln1g, const float* ln1b, const float* ln2g, const float* ln2b,
             const float* exb,
             const _Float16* __restrict__ wpk,
             float* __restrict__ dout, _Float16* __restrict__ tokh, float* __restrict__ pexists) {
  extern __shared__ char smem[];
  _Float16* hbuf  = (_Float16*)smem;          // [64][512] activation / residual stream
  _Float16* accb  = hbuf + 64 * 512;          // [64][512] attn-out accum, then FFN hidden
  _Float16* qbuf  = accb + 64 * 512;          // [64][64]  q, then softmax probs
  _Float16* kbuf  = qbuf + 64 * 64;           // [64][64]  k, then v
  _Float16* pbuf  = kbuf + 64 * 64;           // [64][96]  pts, then z [64][64]
  float*    sc    = (float*)(pbuf + 64 * 96); // [64][64] scores
  float*    validf  = sc + 64 * 64;           // [64]
  float*    avalidf = validf + 64;            // [64]
  int*      flags   = (int*)(avalidf + 64);
  _Float16* wstg  = (_Float16*)(flags + 8);   // [64][512] or [512][64] TDM weight stage (64KB)

  const int n = blockIdx.x, p = n % Pp;
  const int tid = threadIdx.x;

  // ---- build pts = [x | te1 | sin(te_per)] in f16, gather mask ----
  {
    int row = tid >> 2, qq = tid & 3;
    float tval = tt[(size_t)n * Ll + row];
    if (qq == 0) validf[row] = (mask[(size_t)n * Ll + row] > 0) ? 1.0f : 0.0f;
    for (int c = qq * 24; c < qq * 24 + 24; ++c) {
      float v;
      if (c < 16)       v = x[((size_t)n * Ll + row) * DINc + c];
      else if (c == 16) v = tval * tsw[0] + tsb[0];
      else if (c < 80)  v = __sinf(tval * tpw[c - 17] + tpb[c - 17]);
      else              v = 0.0f;
      pbuf[row * DPpad + c] = (_Float16)v;
    }
  }
  if (tid == 0) flags[0] = 0;
  __syncthreads();
  if (tid < 64 && validf[tid] > 0.0f) atomicOr(&flags[0], 1);
  __syncthreads();
  const int has_any = flags[0];
  if (tid < 64) avalidf[tid] = (validf[tid] > 0.0f || (tid == 0 && !has_any)) ? 1.0f : 0.0f;

  // ---- h0 = pts @ inp_w^T + inp_b  (M=64,N=512,K=96) ----
  for (int t = WAVE; t < 128; t += 8) {
    int ti = t >> 5, tj = t & 31;
    v8f c = bcast8(inp_b[tj * 16 + (LANE & 15)]);
    for (int kk = 0; kk < 96; kk += 32)
      c = wmma_f16(ldA16x32(pbuf, ti * 16, kk, DPpad), ldB_nk(wpk + W_INP, tj * 16, kk, DPpad), c);
    stD_f16(hbuf, c, ti * 16, tj * 16, 512);
  }
  __syncthreads();
  if (!has_any) for (int c = tid; c < 512; c += 256) hbuf[c] = (_Float16)0.0f; // zero row 0
  for (int i = tid; i < 64 * 512; i += 256) accb[i] = (_Float16)0.0f;

  // ---- per-head MHA, O-proj accumulated into accb ----
  for (int hd = 0; hd < NHc; ++hd) {
    __syncthreads();
    if (WAVE == 0) {                               // DMA Wq head slice (64x512, contiguous)
      tdm_stage_f16(wpk + W_WQ + (size_t)hd * 64 * 512, wstg, 512, 64, 512);
      tensor_wait0();
    }
    __syncthreads();
    for (int t = WAVE; t < 16; t += 8) {           // q (M=64,N=64,K=512)
      int ti = t >> 2, tj = t & 3, j = hd * 64 + tj * 16 + (LANE & 15);
      v8f cq = bcast8(bq[j]);
      for (int kk = 0; kk < 512; kk += 32)
        cq = wmma_f16(ldA16x32(hbuf, ti * 16, kk, 512), ldB_nk(wstg, tj * 16, kk, 512), cq);
      stD_f16(qbuf, cq, ti * 16, tj * 16, 64);
    }
    __syncthreads();
    if (WAVE == 0) {                               // DMA Wk head slice
      tdm_stage_f16(wpk + W_WK + (size_t)hd * 64 * 512, wstg, 512, 64, 512);
      tensor_wait0();
    }
    __syncthreads();
    for (int t = WAVE; t < 16; t += 8) {           // k
      int ti = t >> 2, tj = t & 3, j = hd * 64 + tj * 16 + (LANE & 15);
      v8f ck = bcast8(bk[j]);
      for (int kk = 0; kk < 512; kk += 32)
        ck = wmma_f16(ldA16x32(hbuf, ti * 16, kk, 512), ldB_nk(wstg, tj * 16, kk, 512), ck);
      stD_f16(kbuf, ck, ti * 16, tj * 16, 64);
    }
    __syncthreads();
    if (WAVE == 0)                                 // DMA Wv, overlapped with scores+softmax
      tdm_stage_f16(wpk + W_WV + (size_t)hd * 64 * 512, wstg, 512, 64, 512);
    for (int t = WAVE; t < 16; t += 8) {           // scores = q@k^T / 8
      int ti = t >> 2, tj = t & 3;
      v8f cs = {};
      for (int kk = 0; kk < 64; kk += 32)
        cs = wmma_f16(ldA16x32(qbuf, ti * 16, kk, 64), ldB_nk(kbuf, tj * 16, kk, 64), cs);
      const int nn = LANE & 15, moff = (LANE >> 4) << 3;
#pragma unroll
      for (int r = 0; r < 8; ++r) sc[(ti * 16 + moff + r) * 64 + tj * 16 + nn] = cs[r] * 0.125f;
    }
    __syncthreads();
    if (tid < 64) {                                // softmax row -> probs f16 in qbuf
      float mx = -1e30f;
      for (int j = 0; j < 64; ++j) if (avalidf[j] > 0.0f) mx = fmaxf(mx, sc[tid * 64 + j]);
      float s = 0.0f;
      for (int j = 0; j < 64; ++j) {
        float e = (avalidf[j] > 0.0f) ? __expf(sc[tid * 64 + j] - mx) : 0.0f;
        sc[tid * 64 + j] = e; s += e;
      }
      float inv = 1.0f / fmaxf(s, 1e-20f);
      for (int j = 0; j < 64; ++j) qbuf[tid * 64 + j] = (_Float16)(sc[tid * 64 + j] * inv);
    }
    __syncthreads();
    if (WAVE == 0) tensor_wait0();                 // Wv landed
    __syncthreads();
    for (int t = WAVE; t < 16; t += 8) {           // v (into kbuf)
      int ti = t >> 2, tj = t & 3, j = hd * 64 + tj * 16 + (LANE & 15);
      v8f cv = bcast8(bv[j]);
      for (int kk = 0; kk < 512; kk += 32)
        cv = wmma_f16(ldA16x32(hbuf, ti * 16, kk, 512), ldB_nk(wstg, tj * 16, kk, 512), cv);
      stD_f16(kbuf, cv, ti * 16, tj * 16, 64);
    }
    __syncthreads();
    if (WAVE == 0)                                 // DMA Wo column slice (strided 2-D tile 64x512)
      tdm_stage_f16(wpk + W_WO + (size_t)hd * 64, wstg, 64, 512, 512);
    for (int t = WAVE; t < 16; t += 8) {           // z = P@V -> pbuf[64][64]
      int ti = t >> 2, tj = t & 3;
      v8f cz = {};
      for (int kk = 0; kk < 64; kk += 32)
        cz = wmma_f16(ldA16x32(qbuf, ti * 16, kk, 64), ldB_kn(kbuf, kk, tj * 16, 64), cz);
      stD_f16(pbuf, cz, ti * 16, tj * 16, 64);
    }
    __syncthreads();
    if (WAVE == 0) tensor_wait0();                 // Wo landed -> staged [512][64], ld=64
    __syncthreads();
    for (int t = WAVE; t < 128; t += 8) {          // accb += z @ wo[:, hd*64:+64]^T
      int ti = t >> 5, tj = t & 31;
      v8f c = ldC_f16(accb, ti * 16, tj * 16, 512);
      for (int kk = 0; kk < 64; kk += 32)
        c = wmma_f16(ldA16x32(pbuf, ti * 16, kk, 64), ldB_nk(wstg, tj * 16, kk, 64), c);
      stD_f16(accb, c, ti * 16, tj * 16, 512);
    }
  }
  __syncthreads();

  // ---- h1 = LN1(h0 + attn_out + bo) ----
  if (tid < 64) {
    float m = 0.0f;
    for (int c = 0; c < 512; ++c) m += (float)hbuf[tid * 512 + c] + (float)accb[tid * 512 + c] + bo[c];
    m *= (1.0f / 512.0f);
    float v = 0.0f;
    for (int c = 0; c < 512; ++c) {
      float d = (float)hbuf[tid * 512 + c] + (float)accb[tid * 512 + c] + bo[c] - m; v += d * d;
    }
    float inv = rsqrtf(v * (1.0f / 512.0f) + 1e-5f);
    for (int c = 0; c < 512; ++c) {
      float d = (float)hbuf[tid * 512 + c] + (float)accb[tid * 512 + c] + bo[c] - m;
      hbuf[tid * 512 + c] = (_Float16)(d * inv * ln1g[c] + ln1b[c]);
    }
  }
  __syncthreads();
  // ---- FFN hidden = relu(h1 @ w1^T + b1) -> accb ----
  for (int t = WAVE; t < 128; t += 8) {
    int ti = t >> 5, tj = t & 31;
    v8f c = bcast8(fb1[tj * 16 + (LANE & 15)]);
    for (int kk = 0; kk < 512; kk += 32)
      c = wmma_f16(ldA16x32(hbuf, ti * 16, kk, 512), ldB_nk(wpk + W_F1, tj * 16, kk, 512), c);
#pragma unroll
    for (int r = 0; r < 8; ++r) c[r] = fmaxf(c[r], 0.0f);
    stD_f16(accb, c, ti * 16, tj * 16, 512);
  }
  __syncthreads();
  // ---- h2pre = h1 + hidden @ w2^T + b2 -> hbuf ----
  for (int t = WAVE; t < 128; t += 8) {
    int ti = t >> 5, tj = t & 31;
    v8f c = ldC_f16(hbuf, ti * 16, tj * 16, 512);
    float bb = fb2[tj * 16 + (LANE & 15)];
#pragma unroll
    for (int r = 0; r < 8; ++r) c[r] += bb;
    for (int kk = 0; kk < 512; kk += 32)
      c = wmma_f16(ldA16x32(accb, ti * 16, kk, 512), ldB_nk(wpk + W_F2, tj * 16, kk, 512), c);
    stD_f16(hbuf, c, ti * 16, tj * 16, 512);
  }
  __syncthreads();
  if (tid < 64) {                                  // LN2 in place
    float m = 0.0f;
    for (int c = 0; c < 512; ++c) m += (float)hbuf[tid * 512 + c];
    m *= (1.0f / 512.0f);
    float v = 0.0f;
    for (int c = 0; c < 512; ++c) { float d = (float)hbuf[tid * 512 + c] - m; v += d * d; }
    float inv = rsqrtf(v * (1.0f / 512.0f) + 1e-5f);
    for (int c = 0; c < 512; ++c) {
      float d = (float)hbuf[tid * 512 + c] - m;
      hbuf[tid * 512 + c] = (_Float16)(d * inv * ln2g[c] + ln2b[c]);
    }
  }
  __syncthreads();
  // ---- masked mean pool + exists bias + positional encoding -> token ----
  float den = 0.0f;
  for (int l = 0; l < 64; ++l) den += validf[l];
  den = fmaxf(den, 1.0f);
  const float pex = has_any ? 1.0f : 0.0f;
  const float lfreq = __logf(10000.0f) / 512.0f;
  for (int j = tid; j < 512; j += 256) {
    float s = 0.0f;
    for (int l = 0; l < 64; ++l) s += validf[l] * (float)hbuf[l * 512 + j];
    float val = s / den + exb[j] * (1.0f - pex);
    float freq = __expf(-(float)(2 * (j >> 1)) * lfreq);
    float ang = (float)p * freq;
    val += (j & 1) ? __cosf(ang) : __sinf(ang);
    dout[(size_t)n * 512 + j] = val;
    tokh[(size_t)n * 512 + j] = (_Float16)val;
  }
  if (tid == 0) pexists[n] = pex;
}

// =============== graph attention: qkv + time-biased attention per (head, batch) ===============
__global__ void __launch_bounds__(256)
gat_qkv_attn(const _Float16* __restrict__ tokh, const _Float16* __restrict__ wpk,
             const float* gbq, const float* gbk, const float* gbv,
             const float* __restrict__ pexists, _Float16* __restrict__ zbuf, int li) {
  extern __shared__ char smem[];
  _Float16* qbuf = (_Float16*)smem;           // [128][64] q, then v
  _Float16* kbuf = qbuf + 128 * 64;           // [128][64]
  _Float16* pbuf = kbuf + 128 * 64;           // [128][128] probs
  float*    sc   = (float*)(pbuf + 128 * 128);// [128][128]
  _Float16* wstg = (_Float16*)(sc + 128 * 128); // [64][512] TDM weight stage
  const int hd = blockIdx.x, b = blockIdx.y, tid = threadIdx.x;
  const _Float16* tok = tokh + (size_t)b * Pp * 512;
  const _Float16* Wq = wpk + W_G + (size_t)(li * 4 + 0) * 512 * 512;
  const _Float16* Wk = wpk + W_G + (size_t)(li * 4 + 1) * 512 * 512;
  const _Float16* Wv = wpk + W_G + (size_t)(li * 4 + 2) * 512 * 512;
  const float* bqv = gbq + li * 512; const float* bkv = gbk + li * 512; const float* bvv = gbv + li * 512;

  if (WAVE == 0) {                                 // DMA Wq head slice
    tdm_stage_f16(Wq + (size_t)hd * 64 * 512, wstg, 512, 64, 512);
    tensor_wait0();
  }
  __syncthreads();
  for (int t = WAVE; t < 32; t += 8) {             // q (M=128,N=64,K=512)
    int ti = t >> 2, tj = t & 3, j = hd * 64 + tj * 16 + (LANE & 15);
    v8f cq = bcast8(bqv[j]);
    for (int kk = 0; kk < 512; kk += 32)
      cq = wmma_f16(ldA16x32(tok, ti * 16, kk, 512), ldB_nk(wstg, tj * 16, kk, 512), cq);
    stD_f16(qbuf, cq, ti * 16, tj * 16, 64);
  }
  __syncthreads();
  if (WAVE == 0) {                                 // DMA Wk head slice
    tdm_stage_f16(Wk + (size_t)hd * 64 * 512, wstg, 512, 64, 512);
    tensor_wait0();
  }
  __syncthreads();
  for (int t = WAVE; t < 32; t += 8) {             // k
    int ti = t >> 2, tj = t & 3, j = hd * 64 + tj * 16 + (LANE & 15);
    v8f ck = bcast8(bkv[j]);
    for (int kk = 0; kk < 512; kk += 32)
      ck = wmma_f16(ldA16x32(tok, ti * 16, kk, 512), ldB_nk(wstg, tj * 16, kk, 512), ck);
    stD_f16(kbuf, ck, ti * 16, tj * 16, 64);
  }
  __syncthreads();
  if (WAVE == 0)                                   // DMA Wv, overlapped with scores+softmax
    tdm_stage_f16(Wv + (size_t)hd * 64 * 512, wstg, 512, 64, 512);
  for (int t = WAVE; t < 64; t += 8) {             // scores + time bias
    int ti = t >> 3, tj = t & 7;
    v8f cs = {};
    for (int kk = 0; kk < 64; kk += 32)
      cs = wmma_f16(ldA16x32(qbuf, ti * 16, kk, 64), ldB_nk(kbuf, tj * 16, kk, 64), cs);
    const int nn = LANE & 15, moff = (LANE >> 4) << 3;
#pragma unroll
    for (int r = 0; r < 8; ++r) {
      int qi = ti * 16 + moff + r, kj = tj * 16 + nn;
      float dist = fabsf((float)(qi - kj)) * DToverTAU;
      float tb = __logf(__expf(-dist) + 1e-12f);
      sc[qi * Pp + kj] = cs[r] * 0.125f + tb;
    }
  }
  __syncthreads();
  if (tid < 128) {                                 // softmax with pexists mask
    const float* pe = pexists + (size_t)b * Pp;
    float mx = -1e30f;
    for (int j = 0; j < 128; ++j) if (pe[j] > 0.0f) mx = fmaxf(mx, sc[tid * Pp + j]);
    float s = 0.0f;
    for (int j = 0; j < 128; ++j) {
      float e = (pe[j] > 0.0f) ? __expf(sc[tid * Pp + j] - mx) : 0.0f;
      sc[tid * Pp + j] = e; s += e;
    }
    float inv = 1.0f / fmaxf(s, 1e-20f);
    for (int j = 0; j < 128; ++j) pbuf[tid * Pp + j] = (_Float16)(sc[tid * Pp + j] * inv);
  }
  __syncthreads();
  if (WAVE == 0) tensor_wait0();                   // Wv landed
  __syncthreads();
  for (int t = WAVE; t < 32; t += 8) {             // v into qbuf (q dead)
    int ti = t >> 2, tj = t & 3, j = hd * 64 + tj * 16 + (LANE & 15);
    v8f cv = bcast8(bvv[j]);
    for (int kk = 0; kk < 512; kk += 32)
      cv = wmma_f16(ldA16x32(tok, ti * 16, kk, 512), ldB_nk(wstg, tj * 16, kk, 512), cv);
    stD_f16(qbuf, cv, ti * 16, tj * 16, 64);
  }
  __syncthreads();
  for (int t = WAVE; t < 32; t += 8) {             // z = P@V -> global z[:, hd*64:+64]
    int ti = t >> 2, tj = t & 3;
    v8f cz = {};
    for (int kk = 0; kk < 128; kk += 32)
      cz = wmma_f16(ldA16x32(pbuf, ti * 16, kk, Pp), ldB_kn(qbuf, kk, tj * 16, 64), cz);
    stD_f16(zbuf + (size_t)(b * Pp) * 512 + hd * 64, cz, ti * 16, tj * 16, 512);
  }
}

// =============== graph attention: O-proj + gate + residual into fp32 tokens ===============
__global__ void __launch_bounds__(256)
gat_oproj(const _Float16* __restrict__ zbuf, const _Float16* __restrict__ wpk,
          const float* gbo, const float* __restrict__ pexists,
          float* __restrict__ dout, _Float16* __restrict__ tokh, int li) {
  const int r0 = blockIdx.x * 64;
  const _Float16* Wo = wpk + W_G + (size_t)(li * 4 + 3) * 512 * 512;
  const float* bov = gbo + li * 512;
  for (int t = WAVE; t < 128; t += 8) {
    int ti = t >> 5, tj = t & 31;
    __builtin_prefetch(Wo + (size_t)(tj * 16) * 512, 0, 0);
    v8f c = bcast8(bov[tj * 16 + (LANE & 15)]);
    for (int kk = 0; kk < 512; kk += 32)
      c = wmma_f16(ldA16x32(zbuf, r0 + ti * 16, kk, 512), ldB_nk(Wo, tj * 16, kk, 512), c);
    const int nn = LANE & 15, moff = (LANE >> 4) << 3;
#pragma unroll
    for (int r = 0; r < 8; ++r) {
      int row = r0 + ti * 16 + moff + r;
      size_t idx = (size_t)row * 512 + tj * 16 + nn;
      float nv = dout[idx] + c[r] * pexists[row];
      dout[idx] = nv;
      tokh[idx] = (_Float16)nv;
    }
  }
}

extern "C" void kernel_launch(void* const* d_in, const int* in_sizes, int n_in,
                              void* d_out, int out_size, void* d_ws, size_t ws_size,
                              hipStream_t stream) {
  const float* x   = (const float*)d_in[0];
  const float* tt  = (const float*)d_in[1];
  const int*  mask = (const int*)d_in[2];
  const float* tsw = (const float*)d_in[3];  const float* tsb = (const float*)d_in[4];
  const float* tpw = (const float*)d_in[5];  const float* tpb = (const float*)d_in[6];
  const float* inp_w = (const float*)d_in[7]; const float* inp_b = (const float*)d_in[8];
  const float* wq = (const float*)d_in[9];   const float* bq = (const float*)d_in[10];
  const float* wk = (const float*)d_in[11];  const float* bk = (const float*)d_in[12];
  const float* wv = (const float*)d_in[13];  const float* bv = (const float*)d_in[14];
  const float* wo = (const float*)d_in[15];  const float* bo = (const float*)d_in[16];
  const float* f1 = (const float*)d_in[17];  const float* fb1 = (const float*)d_in[18];
  const float* f2 = (const float*)d_in[19];  const float* fb2 = (const float*)d_in[20];
  const float* ln1g = (const float*)d_in[21]; const float* ln1b = (const float*)d_in[22];
  const float* ln2g = (const float*)d_in[23]; const float* ln2b = (const float*)d_in[24];
  const float* exb  = (const float*)d_in[25];
  const float* gwq = (const float*)d_in[26]; const float* gbq = (const float*)d_in[27];
  const float* gwk = (const float*)d_in[28]; const float* gbk = (const float*)d_in[29];
  const float* gwv = (const float*)d_in[30]; const float* gbv = (const float*)d_in[31];
  const float* gwo = (const float*)d_in[32]; const float* gbo = (const float*)d_in[33];

  _Float16* wsh  = (_Float16*)d_ws;
  _Float16* tokh = wsh + TOK_OFF;
  _Float16* zb   = wsh + Z_OFF;
  float* pex  = (float*)((char*)d_ws + PEX_OFFB);
  float* dout = (float*)d_out;

  auto pk = [&](const float* s, size_t off, int rows, int cols, int dld) {
    int tot = rows * dld;
    pack_w<<<(tot + 255) / 256, 256, 0, stream>>>(s, wsh + off, rows, cols, dld);
  };
  pk(inp_w, W_INP, 512, 80, 96);
  pk(wq, W_WQ, 512, 512, 512); pk(wk, W_WK, 512, 512, 512);
  pk(wv, W_WV, 512, 512, 512); pk(wo, W_WO, 512, 512, 512);
  pk(f1, W_F1, 512, 512, 512); pk(f2, W_F2, 512, 512, 512);
  for (int li = 0; li < NLc; ++li) {
    pk(gwq + (size_t)li * 512 * 512, W_G + (size_t)(li * 4 + 0) * 512 * 512, 512, 512, 512);
    pk(gwk + (size_t)li * 512 * 512, W_G + (size_t)(li * 4 + 1) * 512 * 512, 512, 512, 512);
    pk(gwv + (size_t)li * 512 * 512, W_G + (size_t)(li * 4 + 2) * 512 * 512, 512, 512, 512);
    pk(gwo + (size_t)li * 512 * 512, W_G + (size_t)(li * 4 + 3) * 512 * 512, 512, 512, 512);
  }

  size_t smem_patch = (size_t)64 * 512 * 2 * 2   // hbuf + accb (f16)
                    + (size_t)64 * 64 * 2 * 2    // qbuf + kbuf
                    + (size_t)64 * 96 * 2        // pbuf
                    + (size_t)64 * 64 * 4        // scores f32
                    + 64 * 4 * 2 + 32            // valid flags
                    + (size_t)64 * 512 * 2;      // TDM weight stage
  patch_kernel<<<Npatch, 256, smem_patch, stream>>>(
      x, tt, mask, tsw, tsb, tpw, tpb, inp_b, bq, bk, bv, bo, fb1, fb2,
      ln1g, ln1b, ln2g, ln2b, exb, wsh, dout, tokh, pex);

  size_t smem_g = (size_t)128 * 64 * 2 * 2 + (size_t)128 * 128 * 2 + (size_t)128 * 128 * 4
                + (size_t)64 * 512 * 2;          // TDM weight stage
  for (int li = 0; li < NLc; ++li) {
    gat_qkv_attn<<<dim3(NHc, Bsz), 256, smem_g, stream>>>(tokh, wsh, gbq, gbk, gbv, pex, zb, li);
    gat_oproj<<<Npatch / 64, 256, 0, stream>>>(zb, wsh, gbo, pex, dout, tokh, li);
  }
}
